// self_multihead_attention_566935683727
// MI455X (gfx1250) — matrix operand order repaired and checked by reading, not compiled, not run on hardware
//
#include <hip/hip_runtime.h>
#include <hip/hip_bf16.h>
#include <math.h>
#include <stdint.h>

typedef __attribute__((ext_vector_type(16))) _Float16 v16h;
typedef __attribute__((ext_vector_type(8)))  _Float16 v8h;
typedef __attribute__((ext_vector_type(8)))  float    v8f;

#define B_DIM 4
#define L_DIM 2048
#define D_DIM 512
#define H_DIM 16
#define HD    32
#define M_TOK (B_DIM * L_DIM)   /* 8192 tokens */

#if __has_builtin(__builtin_amdgcn_global_load_async_to_lds_b128) && \
    __has_builtin(__builtin_amdgcn_s_wait_asynccnt)
#define HAS_ASYNC_LDS 1
// exact parameter types per hipcc diagnostic: int __vector_size__(16) pointers
typedef int v4i_g __attribute__((__vector_size__(4 * sizeof(int))));
typedef __attribute__((address_space(1))) v4i_g* as1_v4i;
typedef __attribute__((address_space(3))) v4i_g* as3_v4i;
#else
#define HAS_ASYNC_LDS 0
#endif

__device__ __forceinline__ v8f wmma16(v16h a, v16h b, v8f c) {
  // D = A(16x32 f16) * B(32x16 f16) + C(16x16 f32)
  return __builtin_amdgcn_wmma_f32_16x16x32_f16(false, a, false, b, (short)0, c,
                                                false, false);
}

// ---------------------------------------------------------------------------
// Elementwise fp32 -> f16 conversion (x)
// ---------------------------------------------------------------------------
__global__ __launch_bounds__(256) void cvt_f32_f16(const float* __restrict__ in,
                                                   _Float16* __restrict__ out,
                                                   int n) {
  int i = blockIdx.x * blockDim.x + threadIdx.x;
  if (i < n) out[i] = (_Float16)in[i];
}

// ---------------------------------------------------------------------------
// Weight transpose + convert: Wt[n][k] = (f16) W[k][n]   (512x512)
// ---------------------------------------------------------------------------
__global__ __launch_bounds__(256) void cvt_w_t(const float* __restrict__ W,
                                               _Float16* __restrict__ Wt) {
  int i = blockIdx.x * blockDim.x + threadIdx.x;   // 0 .. 512*512-1
  int n = i >> 9;
  int k = i & 511;
  Wt[n * D_DIM + k] = (_Float16)W[k * D_DIM + n];
}

// ---------------------------------------------------------------------------
// WMMA GEMM:  out = A(Mx512 f16, row-major) * Bt'(512x512, Bt is N-major) + bias
// One wave computes a 16x64 strip; all 8 waves of a block share the same 64
// weight columns, so the whole 64x512 f16 B strip (64KB) is staged into LDS
// once per block — pre-permuted into WMMA fragment order so inner-loop reads
// are lane-contiguous (bank-conflict-free) ds_load_b128 pairs. Staging uses
// global_load_async_to_lds_b128 (per-lane LDS dest => permute during copy).
// outMode:
//   0: f16, layout (b*16+h, l, d)                (Q / K), value *= scale
//   1: f16, layout (b*16+h, d, l)                (V transposed)
//   2: f32, row-major tokens x 512               (final output)
// ---------------------------------------------------------------------------
__global__ __launch_bounds__(256) void gemm_wmma(
    const _Float16* __restrict__ A, const _Float16* __restrict__ Bt,
    const float* __restrict__ bias, void* __restrict__ out,
    int outMode, float scale) {
  __shared__ _Float16 ldsB[64 * D_DIM];       // 64KB, fragment-ordered
  const int tid   = threadIdx.x;
  const int lane  = tid & 31;
  const int wslot = tid >> 5;                 // 0..7
  const int wid   = blockIdx.x * 8 + wslot;
  const int mtiles = M_TOK >> 4;              // 512
  const int mt = wid % mtiles;
  const int nb = wid / mtiles;                // uniform within a block
  const int m0 = mt << 4;
  const int n0 = nb << 6;

  const int lrow = lane & 15;
  const int asel = (lane & 16) ? 8 : 0;       // A K-chunk select
  const int bsel = (lane & 16) ? 8 : 0;       // B K-chunk select

  // ---- stage B strip into LDS in fragment order -------------------------
  // fragment block (ks, j, c): 32 lanes x 16B at halves [((ks*4+j)*2+c)*256]
#pragma unroll
  for (int s = 0; s < 2; ++s) {
    const int ks = wslot * 2 + s;             // this wave's k-steps
#pragma unroll
    for (int j = 0; j < 4; ++j) {
#pragma unroll
      for (int c = 0; c < 2; ++c) {
        const _Float16* src = Bt + (size_t)(n0 + j * 16 + lrow) * D_DIM +
                              ks * 32 + bsel + c * 16;
        _Float16* dst = &ldsB[(((ks * 4 + j) * 2 + c) << 8) + lane * 8];
#if HAS_ASYNC_LDS
        __builtin_amdgcn_global_load_async_to_lds_b128(
            (as1_v4i)(uintptr_t)src,
            (as3_v4i)(unsigned)(uintptr_t)dst,
            0, 0);
#else
        *(v8h*)dst = *(const v8h*)src;
#endif
      }
    }
  }
#if HAS_ASYNC_LDS
  __builtin_amdgcn_s_wait_asynccnt(0);
#endif
  __syncthreads();

  // ---- main loop --------------------------------------------------------
  const int arow = m0 + lrow;
  v8f acc[4] = {};

  for (int k = 0; k < D_DIM; k += 32) {
    const _Float16* ap = A + (size_t)arow * D_DIM + k + asel;
    v8h a0 = *(const v8h*)(ap);
    v8h a1 = *(const v8h*)(ap + 16);
    v16h Af;
#pragma unroll
    for (int i = 0; i < 8; ++i) { Af[i] = a0[i]; Af[8 + i] = a1[i]; }

    const int ks = k >> 5;
#pragma unroll
    for (int j = 0; j < 4; ++j) {
      const _Float16* b0 = &ldsB[(((ks * 4 + j) * 2 + 0) << 8) + lane * 8];
      const _Float16* b1 = &ldsB[(((ks * 4 + j) * 2 + 1) << 8) + lane * 8];
      v8h bl0 = *(const v8h*)b0;
      v8h bl1 = *(const v8h*)b1;
      v16h Bf;
#pragma unroll
      for (int i = 0; i < 8; ++i) { Bf[i] = bl0[i]; Bf[8 + i] = bl1[i]; }
      acc[j] = wmma16(Af, Bf, acc[j]);
    }
  }

  // ---- epilogue ---------------------------------------------------------
  const int rowAdd = (lane & 16) ? 8 : 0;
#pragma unroll
  for (int j = 0; j < 4; ++j) {
    const int c  = n0 + j * 16 + lrow;
    const float bv = bias[c];
#pragma unroll
    for (int r = 0; r < 8; ++r) {
      const int t = m0 + r + rowAdd;
      float v = (acc[j][r] + bv) * scale;
      if (outMode == 2) {
        ((float*)out)[(size_t)t * D_DIM + c] = v;
      } else {
        const int b = t >> 11, l = t & (L_DIM - 1);
        const int h = c >> 5,  d = c & (HD - 1);
        size_t idx;
        if (outMode == 0)
          idx = ((size_t)(b * H_DIM + h) * L_DIM + l) * HD + d;
        else
          idx = ((size_t)(b * H_DIM + h) * HD + d) * L_DIM + l;
        ((_Float16*)out)[idx] = (_Float16)v;
      }
    }
  }
}

// ---------------------------------------------------------------------------
// Streaming-softmax attention. One wave handles one 16-query tile of one
// (batch, head). Scores computed transposed (M=key, N=query) so exp'ed
// probabilities feed the P*V WMMA A-operand with no cross-lane traffic.
// qh/kh: (bh, l, 32) f16 (Q pre-scaled by 1/sqrt(512)); vth: (bh, 32, l) f16.
// zh out: (token, 512) f16.
// ---------------------------------------------------------------------------
__global__ __launch_bounds__(256) void attn_wmma(
    const _Float16* __restrict__ qh, const _Float16* __restrict__ kh,
    const _Float16* __restrict__ vth, const int* __restrict__ maskp,
    _Float16* __restrict__ zh) {
  const int lane = threadIdx.x & 31;
  const int wid  = (blockIdx.x * blockDim.x + threadIdx.x) >> 5;  // 0..8191
  const int qt   = wid & 127;
  const int bh   = wid >> 7;                 // 0..63
  const int qbase = qt << 4;
  const bool causal = (*maskp) != 0;

  const _Float16* qp = qh + (size_t)bh * L_DIM * HD;
  const _Float16* kp = kh + (size_t)bh * L_DIM * HD;
  const _Float16* vp = vth + (size_t)bh * HD * L_DIM;

  const int asel = (lane & 16) ? 8 : 0;      // K-chunk / row-half select
  const int qcol = lane & 15;
  const int qidx = qbase + qcol;             // this lane's query index

  // Q fragment (B-layout): lane n holds Q[qbase+n][d 0..15 | 16..31]
  v16h Bq = *(const v16h*)(qp + (size_t)(qbase + qcol) * HD +
                           ((lane & 16) ? 16 : 0));

  float mrun = -1e30f, lsum = 0.0f;
  v8f acc0 = {}, acc1 = {};
  const int nblk = causal ? ((qt >> 1) + 1) : (L_DIM / 32);

  for (int kb = 0; kb < nblk * 32; kb += 32) {
    // --- K fragments (A-layout): 16 keys x 32 dims, two subtiles ---
    const _Float16* k0p = kp + (size_t)(kb + qcol) * HD + ((lane & 16) ? 16 : 0);
    v8h a00 = *(const v8h*)(k0p);
    v8h a01 = *(const v8h*)(k0p + 8);
    const _Float16* k1p = k0p + 16 * HD;
    v8h a10 = *(const v8h*)(k1p);
    v8h a11 = *(const v8h*)(k1p + 8);
    v16h A0, A1;
#pragma unroll
    for (int i = 0; i < 8; ++i) {
      A0[i] = a00[i]; A0[8 + i] = a01[i];
      A1[i] = a10[i]; A1[8 + i] = a11[i];
    }

    v8f zc = {};
    v8f st0 = wmma16(A0, Bq, zc);            // S^T keys kb..kb+15
    v8f st1 = wmma16(A1, Bq, zc);            // S^T keys kb+16..kb+31
    const bool valid1 = (!causal) || (kb + 16 <= qbase);

    if (causal && kb == qbase) {             // diagonal subtile 0
#pragma unroll
      for (int r = 0; r < 8; ++r)
        st0[r] = (kb + r + asel > qidx) ? -1e30f : st0[r];
    }
    if (causal && kb + 16 == qbase) {        // diagonal subtile 1
#pragma unroll
      for (int r = 0; r < 8; ++r)
        st1[r] = (kb + 16 + r + asel > qidx) ? -1e30f : st1[r];
    }

    // --- per-query online softmax (lane pair n, n+16 share a query) ---
    float tm = st0[0];
#pragma unroll
    for (int r = 1; r < 8; ++r) tm = fmaxf(tm, st0[r]);
    if (valid1) {
#pragma unroll
      for (int r = 0; r < 8; ++r) tm = fmaxf(tm, st1[r]);
    }
    tm = fmaxf(tm, __shfl_xor(tm, 16, 32));
    const float mnew = fmaxf(mrun, tm);
    const float scl  = __expf(mrun - mnew);

    float p0[8], p1[8], rs = 0.0f;
#pragma unroll
    for (int r = 0; r < 8; ++r) { p0[r] = __expf(st0[r] - mnew); rs += p0[r]; }
    if (valid1) {
#pragma unroll
      for (int r = 0; r < 8; ++r) { p1[r] = __expf(st1[r] - mnew); rs += p1[r]; }
    } else {
#pragma unroll
      for (int r = 0; r < 8; ++r) p1[r] = 0.0f;
    }
    rs += __shfl_xor(rs, 16, 32);
    lsum = lsum * scl + rs;
    mrun = mnew;

    // rescale accumulators: broadcast per-row scale from query-owner lanes
    const int srcBase = (lane & 16) >> 1;    // +8 for upper half rows
#pragma unroll
    for (int r = 0; r < 8; ++r) {
      float sr = __shfl(scl, r + srcBase, 32);
      acc0[r] *= sr;
      acc1[r] *= sr;
    }

    // P fragment (A-layout) — direct from transposed score tiles
    v16h Ap;
#pragma unroll
    for (int r = 0; r < 8; ++r) {
      Ap[r]     = (_Float16)p0[r];
      Ap[8 + r] = (_Float16)p1[r];
    }

    // V fragments (B-layout) from transposed V: two 16B loads (keys asel.., 16+asel..)
    const _Float16* v0p =
        vp + (size_t)qcol * L_DIM + kb + asel;
    v8h v00 = *(const v8h*)(v0p);
    v8h v01 = *(const v8h*)(v0p + 16);
    v8h v10 = *(const v8h*)(v0p + 16 * L_DIM);
    v8h v11 = *(const v8h*)(v0p + 16 * L_DIM + 16);
    v16h Bv0, Bv1;
#pragma unroll
    for (int i = 0; i < 8; ++i) {
      Bv0[i] = v00[i]; Bv0[8 + i] = v01[i];
      Bv1[i] = v10[i]; Bv1[8 + i] = v11[i];
    }

    acc0 = wmma16(Ap, Bv0, acc0);            // Z[:, d 0..15]
    acc1 = wmma16(Ap, Bv1, acc1);            // Z[:, d 16..31]
  }

  // --- normalize + store to (token, 512) f16 ---
  const float inv = 1.0f / lsum;
  const int b = bh >> 4, h = bh & 15;
  const int srcBase = (lane & 16) >> 1;
#pragma unroll
  for (int r = 0; r < 8; ++r) {
    float ir = __shfl(inv, r + srcBase, 32);
    const int t  = b * L_DIM + qbase + r + ((lane & 16) ? 8 : 0);
    const int c0 = h * HD + qcol;
    zh[(size_t)t * D_DIM + c0]      = (_Float16)(acc0[r] * ir);
    zh[(size_t)t * D_DIM + c0 + 16] = (_Float16)(acc1[r] * ir);
  }
}

// ---------------------------------------------------------------------------
extern "C" void kernel_launch(void* const* d_in, const int* in_sizes, int n_in,
                              void* d_out, int out_size, void* d_ws,
                              size_t ws_size, hipStream_t stream) {
  (void)in_sizes; (void)n_in; (void)out_size; (void)ws_size;
  const float* x   = (const float*)d_in[0];
  const float* Wq  = (const float*)d_in[1];
  const float* bq  = (const float*)d_in[2];
  const float* Wk  = (const float*)d_in[3];
  const float* bk  = (const float*)d_in[4];
  const float* Wv  = (const float*)d_in[5];
  const float* bv  = (const float*)d_in[6];
  const float* Wf  = (const float*)d_in[7];
  const float* bf  = (const float*)d_in[8];
  const int* maskp = (const int*)d_in[9];

  // workspace carve-up (~42 MB)
  _Float16* xh  = (_Float16*)d_ws;
  _Float16* wqt = xh  + (size_t)M_TOK * D_DIM;
  _Float16* wkt = wqt + (size_t)D_DIM * D_DIM;
  _Float16* wvt = wkt + (size_t)D_DIM * D_DIM;
  _Float16* wft = wvt + (size_t)D_DIM * D_DIM;
  _Float16* qh  = wft + (size_t)D_DIM * D_DIM;
  _Float16* kh  = qh  + (size_t)M_TOK * D_DIM;
  _Float16* vth = kh  + (size_t)M_TOK * D_DIM;
  _Float16* zh  = vth + (size_t)M_TOK * D_DIM;

  const int nx = M_TOK * D_DIM;                       // 4,194,304
  cvt_f32_f16<<<(nx + 255) / 256, 256, 0, stream>>>(x, xh, nx);
  cvt_w_t<<<(D_DIM * D_DIM) / 256, 256, 0, stream>>>(Wq, wqt);
  cvt_w_t<<<(D_DIM * D_DIM) / 256, 256, 0, stream>>>(Wk, wkt);
  cvt_w_t<<<(D_DIM * D_DIM) / 256, 256, 0, stream>>>(Wv, wvt);
  cvt_w_t<<<(D_DIM * D_DIM) / 256, 256, 0, stream>>>(Wf, wft);

  const int gemmBlocks = (M_TOK / 16) * 8 * 32 / 256; // 512
  const float qscale = 0.044194173824159216f;         // 1/sqrt(512)
  gemm_wmma<<<gemmBlocks, 256, 0, stream>>>(xh, wqt, bq, qh, 0, qscale);
  gemm_wmma<<<gemmBlocks, 256, 0, stream>>>(xh, wkt, bk, kh, 0, 1.0f);
  gemm_wmma<<<gemmBlocks, 256, 0, stream>>>(xh, wvt, bv, vth, 1, 1.0f);

  // 8192 query tiles, one wave each -> 1024 blocks of 8 waves
  attn_wmma<<<1024, 256, 0, stream>>>(qh, kh, vth, maskp, zh);

  gemm_wmma<<<gemmBlocks, 256, 0, stream>>>(zh, wft, bf, d_out, 2, 1.0f);
}
